// InvariantGNN_23871428231324
// MI455X (gfx1250) — matrix operand order repaired and checked by reading, not compiled, not run on hardware
//
#include <hip/hip_runtime.h>
#include <hip/hip_bf16.h>

// ---------------------------------------------------------------------------
// InvariantGNN on MI455X (gfx1250): bf16 WMMA edge-MLP message passing,
// async Global->LDS gather staging, f32-atomic segment sum.
// ---------------------------------------------------------------------------

#define N_NODES   100000
#define N_EDGES   1000000
#define NODE_DIM  64
#define EDGE_DIM  32
#define N_LAYERS  4
#define NUM_GAUSS 16
#define HIDDEN    128
#define CAT_DIM   160   // 2*NODE_DIM + EDGE_DIM
#define NUM_GRAPHS 64

typedef __attribute__((ext_vector_type(16))) __bf16 v16bf;
typedef __attribute__((ext_vector_type(8)))  float  v8f;

union FragAB { uint4 u[2]; v16bf v; };

#if __has_builtin(__builtin_amdgcn_global_load_async_to_lds_b128)
#define HAVE_ASYNC_LDS 1
typedef int v4i_async __attribute__((vector_size(16)));
typedef __attribute__((address_space(1))) v4i_async Gv4i;
typedef __attribute__((address_space(3))) v4i_async Lv4i;
#else
#define HAVE_ASYNC_LDS 0
#endif

__device__ __forceinline__ void wait_asynccnt0() {
#if HAVE_ASYNC_LDS
#if __has_builtin(__builtin_amdgcn_s_wait_asynccnt)
    __builtin_amdgcn_s_wait_asynccnt(0);
#else
    asm volatile("s_wait_asynccnt 0" ::: "memory");
#endif
#endif
}

__device__ __forceinline__ unsigned short f2bf(float x) {
    unsigned int u = __float_as_uint(x);
    unsigned int r = (u + 0x7FFFu + ((u >> 16) & 1u)) >> 16;   // RNE
    return (unsigned short)r;
}

// fast SiLU: x * rcp(1+exp(-x)) -> v_exp_f32 + v_rcp_f32 + v_mul_f32
__device__ __forceinline__ float fast_silu(float x) {
    return x * __builtin_amdgcn_rcpf(1.0f + __expf(-x));
}

// ------------------------------ prep kernels -------------------------------

__global__ void zero_f32_kernel(float* __restrict__ p, int n) {
    int i = blockIdx.x * blockDim.x + threadIdx.x;
    if (i < n) p[i] = 0.0f;
}

__global__ void embed_kernel(const int* __restrict__ atoms,
                             const float* __restrict__ emb,
                             float* __restrict__ h,
                             unsigned short* __restrict__ hbf) {
    int i = blockIdx.x * blockDim.x + threadIdx.x;
    if (i >= N_NODES * NODE_DIM) return;
    int n = i >> 6, d = i & 63;
    float v = emb[atoms[n] * NODE_DIM + d];
    h[i] = v;
    hbf[i] = f2bf(v);
}

__global__ void edge_attr_kernel(const int* __restrict__ srcI,
                                 const int* __restrict__ dstI,
                                 const float* __restrict__ coord,
                                 const int* __restrict__ isrec,
                                 const float* __restrict__ subw,
                                 unsigned short* __restrict__ ea) {
    int e = blockIdx.x * blockDim.x + threadIdx.x;
    if (e >= N_EDGES) return;
    int s = srcI[e], d = dstI[e];
    float dx = coord[s * 3 + 0] - coord[d * 3 + 0];
    float dy = coord[s * 3 + 1] - coord[d * 3 + 1];
    float dz = coord[s * 3 + 2] - coord[d * 3 + 2];
    float dist = sqrtf(dx * dx + dy * dy + dz * dz);
    const float step  = 5.0f / (float)(NUM_GAUSS - 1);
    const float coeff = -0.5f / (step * step);
    unsigned short* row = ea + (size_t)e * EDGE_DIM;
#pragma unroll
    for (int g = 0; g < NUM_GAUSS; g++) {
        float off = step * (float)g;
        float t = dist - off;
        row[g] = f2bf(__expf(coeff * t * t));
    }
    int kind = (isrec[s] != isrec[d]) ? 1 : 0;
#pragma unroll
    for (int j = 0; j < NUM_GAUSS; j++)
        row[NUM_GAUSS + j] = f2bf(subw[kind * NUM_GAUSS + j]);
}

// Map (lane, i) within a WMMA-B bf16 fragment to (k, n) offsets (ISA 7.12.2).
__device__ __forceinline__ void b_frag_kn(int lane, int i, int& kOff, int& nOff) {
    int half  = lane >> 4;
    int laneN = lane & 15;
    int v = i >> 1, pos = i & 1;
    // B (KxN): element i, lanes 0-15 -> K = i (i<8) / 8+i (i>=8), lanes 16-31 -> K = 8+i (i<8) / 16+i (i>=8)
    kOff = half * 8 + (i & 7) + (i >> 3) * 16;
    nOff = laneN;
}

__global__ void pack_w1_kernel(const float* __restrict__ W1,
                               unsigned short* __restrict__ w1p) {
    int i = blockIdx.x * blockDim.x + threadIdx.x;
    const int total = N_LAYERS * 5 * 8 * 512;       // 5 K-tiles, 8 N-tiles
    if (i >= total) return;
    int idx  = i & 511;
    int frag = i >> 9;
    int nt = frag & 7;
    int kt = (frag >> 3) % 5;
    int l  = frag / 40;
    int lane = idx >> 4, ii = idx & 15;
    int kOff, nOff;
    b_frag_kn(lane, ii, kOff, nOff);
    int k = kt * 32 + kOff;
    int n = nt * 16 + nOff;
    w1p[i] = f2bf(W1[((size_t)l * CAT_DIM + k) * HIDDEN + n]);
}

__global__ void pack_w2_kernel(const float* __restrict__ W2,
                               unsigned short* __restrict__ w2p) {
    int i = blockIdx.x * blockDim.x + threadIdx.x;
    const int total = N_LAYERS * 4 * 4 * 512;       // 4 K-tiles, 4 N-tiles
    if (i >= total) return;
    int idx  = i & 511;
    int frag = i >> 9;
    int nt = frag & 3;
    int kt = (frag >> 2) & 3;
    int l  = frag >> 4;
    int lane = idx >> 4, ii = idx & 15;
    int kOff, nOff;
    b_frag_kn(lane, ii, kOff, nOff);
    int k = kt * 32 + kOff;
    int n = nt * 16 + nOff;
    w2p[i] = f2bf(W2[((size_t)l * HIDDEN + k) * NODE_DIM + n]);
}

// --------------------------- WMMA edge-MLP kernel --------------------------
// One wave per 16-edge tile. m = [h[dst] | h[src] | edge_attr]  (16 x 160 bf16)
// hidden = silu(m @ W1 + b1)  (16 x 128), out = hidden @ W2 + b2 (16 x 64),
// atomically accumulated into agg[dst].

#define WAVES_PB 4

__global__ __launch_bounds__(32 * WAVES_PB)
void edge_mlp_kernel(const unsigned short* __restrict__ hbf,
                     const unsigned short* __restrict__ ea,
                     const unsigned short* __restrict__ w1p,
                     const unsigned short* __restrict__ w2p,
                     const float* __restrict__ b1,
                     const float* __restrict__ b2,
                     const int* __restrict__ srcI,
                     const int* __restrict__ dstI,
                     float* __restrict__ agg,
                     int layer) {
    __shared__ unsigned short mTile[WAVES_PB][16 * CAT_DIM];
    __shared__ unsigned short hTile[WAVES_PB][16 * HIDDEN];
    __shared__ int dTile[WAVES_PB][16];

    const int wave = threadIdx.x >> 5;
    const int lane = threadIdx.x & 31;
    const int tile = blockIdx.x * WAVES_PB + wave;      // exactly E/16 tiles
    const int ebase = tile * 16;

    unsigned short* m  = mTile[wave];
    unsigned short* hh = hTile[wave];
    int* dd = dTile[wave];

    // ---- stage m tile: lanes 0-15 copy h[dst] row, lanes 16-31 copy h[src]
    //      row + edge_attr row. Use async Global->LDS DMA when available. ----
    {
        int row = lane & 15;
        int e = ebase + row;
        if (lane < 16) {
            int d = dstI[e];
            dd[row] = d;
            const unsigned short* gp = hbf + (size_t)d * NODE_DIM;
            unsigned short* lp = m + row * CAT_DIM;
#if HAVE_ASYNC_LDS
#pragma unroll
            for (int i = 0; i < 8; i++)
                __builtin_amdgcn_global_load_async_to_lds_b128(
                    (Gv4i*)(gp + i * 8), (Lv4i*)(lp + i * 8), 0, 0);
#else
#pragma unroll
            for (int i = 0; i < 8; i++)
                ((uint4*)lp)[i] = ((const uint4*)gp)[i];
#endif
        } else {
            int s = srcI[e];
            const unsigned short* gp = hbf + (size_t)s * NODE_DIM;
            unsigned short* lp = m + row * CAT_DIM + NODE_DIM;
            const unsigned short* ap = ea + (size_t)e * EDGE_DIM;
            unsigned short* alp = m + row * CAT_DIM + 2 * NODE_DIM;
#if HAVE_ASYNC_LDS
#pragma unroll
            for (int i = 0; i < 8; i++)
                __builtin_amdgcn_global_load_async_to_lds_b128(
                    (Gv4i*)(gp + i * 8), (Lv4i*)(lp + i * 8), 0, 0);
#pragma unroll
            for (int i = 0; i < 4; i++)
                __builtin_amdgcn_global_load_async_to_lds_b128(
                    (Gv4i*)(ap + i * 8), (Lv4i*)(alp + i * 8), 0, 0);
#else
#pragma unroll
            for (int i = 0; i < 8; i++)
                ((uint4*)lp)[i] = ((const uint4*)gp)[i];
#pragma unroll
            for (int i = 0; i < 4; i++)
                ((uint4*)alp)[i] = ((const uint4*)ap)[i];
#endif
        }
    }
    wait_asynccnt0();
    __syncthreads();

    const int row   = lane & 15;
    const int kHalf = (lane >= 16) ? 8 : 0;   // A-frag: hi lanes hold K+8..15 / K+24..31
    const int rbase = (lane >= 16) ? 8 : 0;   // C/D-frag: hi lanes hold rows 8..15

    // ---------------- GEMM 1: hidden = silu(m @ W1 + b1) ----------------
    for (int nt = 0; nt < 8; nt++) {
        float bias = b1[layer * HIDDEN + nt * 16 + row];
        v8f acc;
#pragma unroll
        for (int j = 0; j < 8; j++) acc[j] = bias;
#pragma unroll
        for (int kt = 0; kt < 5; kt++) {
            int kb = kt * 32 + kHalf;
            FragAB A, B;
            A.u[0] = *(const uint4*)(m + row * CAT_DIM + kb);
            A.u[1] = *(const uint4*)(m + row * CAT_DIM + kb + 16);
            size_t frag = ((size_t)(layer * 5 + kt) * 8 + nt);
            const uint4* bp = (const uint4*)(w1p + frag * 512 + lane * 16);
            B.u[0] = bp[0];
            B.u[1] = bp[1];
            acc = __builtin_amdgcn_wmma_f32_16x16x32_bf16(
                false, A.v, false, B.v, (short)0, acc, false, false);
        }
        int col = nt * 16 + row;
#pragma unroll
        for (int j = 0; j < 8; j++) {
            hh[(rbase + j) * HIDDEN + col] = f2bf(fast_silu(acc[j]));
        }
    }
    __syncthreads();

    // ---------------- GEMM 2: out = hidden @ W2 + b2 -> agg[dst] --------
    for (int nt = 0; nt < 4; nt++) {
        float bias = b2[layer * NODE_DIM + nt * 16 + row];
        v8f acc;
#pragma unroll
        for (int j = 0; j < 8; j++) acc[j] = bias;
#pragma unroll
        for (int kt = 0; kt < 4; kt++) {
            int kb = kt * 32 + kHalf;
            FragAB A, B;
            A.u[0] = *(const uint4*)(hh + row * HIDDEN + kb);
            A.u[1] = *(const uint4*)(hh + row * HIDDEN + kb + 16);
            size_t frag = ((size_t)(layer * 4 + kt) * 4 + nt);
            const uint4* bp = (const uint4*)(w2p + frag * 512 + lane * 16);
            B.u[0] = bp[0];
            B.u[1] = bp[1];
            acc = __builtin_amdgcn_wmma_f32_16x16x32_bf16(
                false, A.v, false, B.v, (short)0, acc, false, false);
        }
        int col = nt * 16 + row;
#pragma unroll
        for (int j = 0; j < 8; j++) {
            int r = rbase + j;
            atomicAdd(agg + (size_t)dd[r] * NODE_DIM + col, acc[j]);
        }
    }
}

// --------------------------- epilogue kernels ------------------------------

__global__ void update_h_kernel(float* __restrict__ h,
                                const float* __restrict__ agg,
                                unsigned short* __restrict__ hbf) {
    int i = blockIdx.x * blockDim.x + threadIdx.x;
    if (i >= N_NODES * NODE_DIM) return;
    float v = h[i] + agg[i];
    v = v > 0.0f ? v : 0.0f;      // ReLU
    h[i] = v;
    hbf[i] = f2bf(v);
}

__global__ void pool_kernel(const float* __restrict__ h,
                            const int* __restrict__ batch,
                            float* __restrict__ pooled,
                            float* __restrict__ counts) {
    int i = blockIdx.x * blockDim.x + threadIdx.x;
    if (i >= N_NODES * NODE_DIM) return;
    int n = i >> 6, d = i & 63;
    int g = batch[n];
    atomicAdd(&pooled[g * NODE_DIM + d], h[i]);
    if (d == 0) atomicAdd(&counts[g], 1.0f);
}

__global__ void out_kernel(const float* __restrict__ pooled,
                           const float* __restrict__ counts,
                           const float* __restrict__ fcw,
                           const float* __restrict__ fcb,
                           float* __restrict__ out) {
    int g = threadIdx.x;
    if (g >= NUM_GRAPHS) return;
    float c = counts[g];
    c = c < 1.0f ? 1.0f : c;
    float inv = __builtin_amdgcn_rcpf(c);
    float acc = 0.0f;
#pragma unroll
    for (int d = 0; d < NODE_DIM; d++)
        acc += pooled[g * NODE_DIM + d] * fcw[d];
    out[g] = acc * inv + fcb[0];
}

// ------------------------------- launcher ----------------------------------

extern "C" void kernel_launch(void* const* d_in, const int* in_sizes, int n_in,
                              void* d_out, int out_size, void* d_ws, size_t ws_size,
                              hipStream_t stream) {
    const int*   atoms   = (const int*)d_in[0];
    const int*   eidx    = (const int*)d_in[1];        // [2, E]
    const float* coord   = (const float*)d_in[2];
    const int*   isrec   = (const int*)d_in[3];
    const int*   batch   = (const int*)d_in[4];
    const float* nemb    = (const float*)d_in[5];
    const float* subw    = (const float*)d_in[6];
    const float* W1      = (const float*)d_in[7];
    const float* b1      = (const float*)d_in[8];
    const float* W2      = (const float*)d_in[9];
    const float* b2      = (const float*)d_in[10];
    const float* fcw     = (const float*)d_in[11];
    const float* fcb     = (const float*)d_in[12];
    float* out = (float*)d_out;

    const int* srcI = eidx;
    const int* dstI = eidx + N_EDGES;

    // workspace carve-up (256B aligned)
    char* ws = (char*)d_ws;
    size_t off = 0;
    auto carve = [&](size_t bytes) {
        char* p = ws + off;
        off = (off + bytes + 255) & ~(size_t)255;
        return p;
    };
    float*          h_f32  = (float*)carve((size_t)N_NODES * NODE_DIM * 4);
    unsigned short* h_bf   = (unsigned short*)carve((size_t)N_NODES * NODE_DIM * 2);
    float*          agg    = (float*)carve((size_t)N_NODES * NODE_DIM * 4);
    unsigned short* ea     = (unsigned short*)carve((size_t)N_EDGES * EDGE_DIM * 2);
    unsigned short* w1p    = (unsigned short*)carve((size_t)N_LAYERS * 5 * 8 * 512 * 2);
    unsigned short* w2p    = (unsigned short*)carve((size_t)N_LAYERS * 4 * 4 * 512 * 2);
    float*          pooled = (float*)carve((size_t)NUM_GRAPHS * NODE_DIM * 4 +
                                           (size_t)NUM_GRAPHS * 4);
    float*          counts = pooled + NUM_GRAPHS * NODE_DIM;

    const int TPB = 256;
    const int nh  = N_NODES * NODE_DIM;          // 6.4M

    // prep
    pack_w1_kernel<<<(N_LAYERS * 5 * 8 * 512 + TPB - 1) / TPB, TPB, 0, stream>>>(W1, w1p);
    pack_w2_kernel<<<(N_LAYERS * 4 * 4 * 512 + TPB - 1) / TPB, TPB, 0, stream>>>(W2, w2p);
    embed_kernel<<<(nh + TPB - 1) / TPB, TPB, 0, stream>>>(atoms, nemb, h_f32, h_bf);
    edge_attr_kernel<<<(N_EDGES + TPB - 1) / TPB, TPB, 0, stream>>>(
        srcI, dstI, coord, isrec, subw, ea);

    // message passing layers
    const int tiles = N_EDGES / 16;              // 62500
    const int mlpBlocks = tiles / WAVES_PB;      // 15625 (exact)
    for (int l = 0; l < N_LAYERS; l++) {
        zero_f32_kernel<<<(nh + TPB - 1) / TPB, TPB, 0, stream>>>(agg, nh);
        edge_mlp_kernel<<<mlpBlocks, 32 * WAVES_PB, 0, stream>>>(
            h_bf, ea, w1p, w2p, b1, b2, srcI, dstI, agg, l);
        update_h_kernel<<<(nh + TPB - 1) / TPB, TPB, 0, stream>>>(h_f32, agg, h_bf);
    }

    // pooling + final FC
    const int poolN = NUM_GRAPHS * NODE_DIM + NUM_GRAPHS;
    zero_f32_kernel<<<(poolN + TPB - 1) / TPB, TPB, 0, stream>>>(pooled, poolN);
    pool_kernel<<<(nh + TPB - 1) / TPB, TPB, 0, stream>>>(h_f32, batch, pooled, counts);
    out_kernel<<<1, 64, 0, stream>>>(pooled, counts, fcw, fcb, out);
}
